// ODERNN_57578331570467
// MI455X (gfx1250) — matrix operand order repaired and checked by reading, not compiled, not run on hardware
//
#include <hip/hip_runtime.h>
#include <hip/hip_bf16.h>

typedef __attribute__((ext_vector_type(16))) _Float16 v16h;
typedef __attribute__((ext_vector_type(8)))  _Float16 v8h;
typedef __attribute__((ext_vector_type(8)))  float    v8f;

#define B_  256
#define T_  64
#define C_  64
#define H_  256
#define HH_ 1024
#define O_  32

__device__ __forceinline__ float sigmoidf_(float x) { return 1.0f / (1.0f + expf(-x)); }

// ---------------------------------------------------------------------------
// fp32 -> f16 conversion (weights / x), once per launch
// ---------------------------------------------------------------------------
__global__ void k_cvt_f32_f16(const float* __restrict__ s, _Float16* __restrict__ d, int n) {
    int i = blockIdx.x * blockDim.x + threadIdx.x;
    if (i < n) d[i] = (_Float16)s[i];
}

__global__ void k_init_h(float* __restrict__ h, _Float16* __restrict__ hz, int n) {
    int i = blockIdx.x * blockDim.x + threadIdx.x;
    if (i < n) { h[i] = 0.0f; hz[i] = (_Float16)0.0f; }
}

// ---------------------------------------------------------------------------
// Fragment loaders (16-bit A 16x32 / B 32x16, contiguous k = 16*half + s, wave32)
// ---------------------------------------------------------------------------
__device__ __forceinline__ v16h load_a_frag(const _Float16* __restrict__ Ap, int k0) {
    v8h alo = *(const v8h*)(Ap + k0);
    v8h ahi = *(const v8h*)(Ap + k0 + 8);
    v16h av;
#pragma unroll
    for (int i = 0; i < 8; ++i) { av[i] = alo[i]; av[i + 8] = ahi[i]; }
    return av;
}

// ---------------------------------------------------------------------------
// Generic WMMA GEMM with tanh + f16 output (lin1 / lin2):
//   Out[M x N](f16) = tanh( A[M x K] @ W[N x K]^T + bias[N] )
// One wave -> 16x64 strip (4 tiles, A reused 4x); 8 waves/block.
// ---------------------------------------------------------------------------
__global__ __launch_bounds__(256) void gemm_tanh(
    const _Float16* __restrict__ A, int lda,
    const _Float16* __restrict__ W,
    const float* __restrict__ bias,
    _Float16* __restrict__ Out, int ldo,
    int M, int N, int K)
{
    const int lane = threadIdx.x & 31;
    const int wave = threadIdx.x >> 5;
    const int ngrp = N >> 6;
    const int wt   = blockIdx.x * 8 + wave;
    if (wt >= (M >> 4) * ngrp) return;             // wave-uniform
    const int tm = wt / ngrp, tn = wt % ngrp;
    const int m0 = tm << 4, n0 = tn << 6;
    const int half = lane >> 4, l15 = lane & 15;

    const _Float16* Ap = A + (size_t)(m0 + l15) * lda + half * 16;
    const _Float16* Wp = W + (size_t)(n0 + l15) * K + half * 16;
    const size_t wstride = (size_t)16 * K;

    v8f acc[4] = {};
    for (int k0 = 0; k0 < K; k0 += 32) {
        v16h av = load_a_frag(Ap, k0);
#pragma unroll
        for (int j = 0; j < 4; ++j) {
            v16h bv = *(const v16h*)(Wp + (size_t)j * wstride + k0);
            acc[j] = __builtin_amdgcn_wmma_f32_16x16x32_f16(
                false, av, false, bv, (short)0, acc[j], false, false);
        }
    }

    const int rbase = m0 + half * 8;
#pragma unroll
    for (int j = 0; j < 4; ++j) {
        const int col = n0 + j * 16 + l15;
        const float bb = bias[col];
#pragma unroll
        for (int r = 0; r < 8; ++r) {
            float v = tanhf(acc[j][r] + bb);
            Out[(size_t)(rbase + r) * ldo + col] = (_Float16)v;
        }
    }
}

// ---------------------------------------------------------------------------
// lin3 with fused RK4 (3/8-rule) combine in the epilogue.
//   kv = A @ w3^T + b3           (the current k_i, at [B x H])
//   MODE 1: k1=kv;  z = y + dt*kv/3                  -> zout (f16)
//   MODE 2: k2=kv;  z = y + dt*(kv - k1/3)           -> zout (f16)
//   MODE 3: k3=kv;  z = y + dt*(k1 - k2 + kv)        -> zout (f16)
//   MODE 4: y += (k1 + 3(k2+k3) + kv)*dt/8 ; y->f32, zout = f16(y)
// ---------------------------------------------------------------------------
template<int MODE>
__global__ __launch_bounds__(256) void gemm_lin3(
    const _Float16* __restrict__ A,          // [B x HH] f16 (a2)
    const _Float16* __restrict__ W,          // [H x HH] f16 (w3)
    const float* __restrict__ bias,          // b3
    float* __restrict__ y,                   // f32 state (read; MODE4: write)
    float* __restrict__ k1, float* __restrict__ k2, float* __restrict__ k3,
    _Float16* __restrict__ zout,             // next RK4 arg (f16) or new h (f16)
    const float* __restrict__ times, int t)
{
    const int lane = threadIdx.x & 31;
    const int wave = threadIdx.x >> 5;
    const int ngrp = H_ >> 6;                      // 4
    const int wt   = blockIdx.x * 8 + wave;
    if (wt >= (B_ >> 4) * ngrp) return;
    const int tm = wt / ngrp, tn = wt % ngrp;
    const int m0 = tm << 4, n0 = tn << 6;
    const int half = lane >> 4, l15 = lane & 15;

    const _Float16* Ap = A + (size_t)(m0 + l15) * HH_ + half * 16;
    const _Float16* Wp = W + (size_t)(n0 + l15) * HH_ + half * 16;
    const size_t wstride = (size_t)16 * HH_;

    v8f acc[4] = {};
    for (int k0 = 0; k0 < HH_; k0 += 32) {
        v16h av = load_a_frag(Ap, k0);
#pragma unroll
        for (int j = 0; j < 4; ++j) {
            v16h bv = *(const v16h*)(Wp + (size_t)j * wstride + k0);
            acc[j] = __builtin_amdgcn_wmma_f32_16x16x32_f16(
                false, av, false, bv, (short)0, acc[j], false, false);
        }
    }

    const float dt = times[t] - times[t - 1];
    const int rbase = m0 + half * 8;
#pragma unroll
    for (int j = 0; j < 4; ++j) {
        const int col = n0 + j * 16 + l15;
        const float bb = bias[col];
#pragma unroll
        for (int r = 0; r < 8; ++r) {
            const size_t idx = (size_t)(rbase + r) * H_ + col;
            const float kv = acc[j][r] + bb;
            if (MODE == 1) {
                k1[idx] = kv;
                zout[idx] = (_Float16)(y[idx] + dt * kv * (1.0f / 3.0f));
            } else if (MODE == 2) {
                k2[idx] = kv;
                zout[idx] = (_Float16)(y[idx] + dt * (kv - k1[idx] * (1.0f / 3.0f)));
            } else if (MODE == 3) {
                k3[idx] = kv;
                zout[idx] = (_Float16)(y[idx] + dt * (k1[idx] - k2[idx] + kv));
            } else {
                const float yn = y[idx] +
                    (k1[idx] + 3.0f * (k2[idx] + k3[idx]) + kv) * dt * 0.125f;
                y[idx]    = yn;
                zout[idx] = (_Float16)yn;
            }
        }
    }
}

// ---------------------------------------------------------------------------
// Fully fused GRU cell: one wave owns a 16x16 (rows, j) tile and computes all
// six gate tiles (r/z/n from x@W_ih^T and h@W_hh^T) in 6 accumulators, then
// does the gate math in-register.
//   hin16 : f16 copy of the incoming hidden state (A operand of gh GEMMs)
//   y     : f32 incoming hidden state (blended); overwritten with new h
//   hout16: f16 copy of the NEW hidden state (separate buffer -> no RAW hazard)
// ---------------------------------------------------------------------------
__global__ __launch_bounds__(256) void gru_fused(
    const _Float16* __restrict__ Xt, int ldx,    // xh + t*C, lda = T*C
    const _Float16* __restrict__ Wih,            // [3H x C] f16
    const _Float16* __restrict__ Whh,            // [3H x H] f16
    const float* __restrict__ b_ih, const float* __restrict__ b_hh,
    const _Float16* __restrict__ hin16,
    float* __restrict__ y, _Float16* __restrict__ hout16)
{
    const int lane = threadIdx.x & 31;
    const int wave = threadIdx.x >> 5;
    const int wt   = blockIdx.x * 8 + wave;
    if (wt >= (B_ >> 4) * (H_ >> 4)) return;      // 16 * 16 = 256 waves
    const int tm = wt >> 4, tn = wt & 15;
    const int m0 = tm << 4, j0 = tn << 4;
    const int half = lane >> 4, l15 = lane & 15;

    v8f ai[3] = {};
    v8f ah[3] = {};

    // gi = x_t @ W_ih^T  (K = C = 64)
    {
        const _Float16* Ap = Xt + (size_t)(m0 + l15) * ldx + half * 16;
#pragma unroll
        for (int k0 = 0; k0 < C_; k0 += 32) {
            v16h av = load_a_frag(Ap, k0);
#pragma unroll
            for (int g = 0; g < 3; ++g) {
                v16h bv = *(const v16h*)(Wih + (size_t)(g * H_ + j0 + l15) * C_ + half * 16 + k0);
                ai[g] = __builtin_amdgcn_wmma_f32_16x16x32_f16(
                    false, av, false, bv, (short)0, ai[g], false, false);
            }
        }
    }
    // gh = h @ W_hh^T  (K = H = 256)
    {
        const _Float16* Ap = hin16 + (size_t)(m0 + l15) * H_ + half * 16;
        for (int k0 = 0; k0 < H_; k0 += 32) {
            v16h av = load_a_frag(Ap, k0);
#pragma unroll
            for (int g = 0; g < 3; ++g) {
                v16h bv = *(const v16h*)(Whh + (size_t)(g * H_ + j0 + l15) * H_ + half * 16 + k0);
                ah[g] = __builtin_amdgcn_wmma_f32_16x16x32_f16(
                    false, av, false, bv, (short)0, ah[g], false, false);
            }
        }
    }

    const int col = j0 + l15;
    const float bi0 = b_ih[col], bi1 = b_ih[H_ + col], bi2 = b_ih[2 * H_ + col];
    const float bh0 = b_hh[col], bh1 = b_hh[H_ + col], bh2 = b_hh[2 * H_ + col];
    const int rbase = m0 + half * 8;
#pragma unroll
    for (int r = 0; r < 8; ++r) {
        const size_t idx = (size_t)(rbase + r) * H_ + col;
        const float ir = ai[0][r] + bi0, iz = ai[1][r] + bi1, in_ = ai[2][r] + bi2;
        const float hr = ah[0][r] + bh0, hz = ah[1][r] + bh1, hn  = ah[2][r] + bh2;
        const float rg = sigmoidf_(ir + hr);
        const float zg = sigmoidf_(iz + hz);
        const float ng = tanhf(in_ + rg * hn);
        const float hv = (1.0f - zg) * ng + zg * y[idx];
        y[idx]      = hv;
        hout16[idx] = (_Float16)hv;
    }
}

// ---------------------------------------------------------------------------
// Output head + softmax: one wave per batch row, lane = output class
// ---------------------------------------------------------------------------
__global__ __launch_bounds__(32) void k_head(const float* __restrict__ h,
                                             const float* __restrict__ Wo,
                                             const float* __restrict__ bo,
                                             float* __restrict__ out)
{
    const int b = blockIdx.x;
    const int o = threadIdx.x;               // 0..31
    float acc = bo[o];
    const float* hr = h + (size_t)b * H_;
    const float* wr = Wo + (size_t)o * H_;
    for (int k = 0; k < H_; ++k) acc += hr[k] * wr[k];
    float m = acc;
    for (int off = 16; off; off >>= 1) m = fmaxf(m, __shfl_xor(m, off, 32));
    const float e = expf(acc - m);
    float s = e;
    for (int off = 16; off; off >>= 1) s += __shfl_xor(s, off, 32);
    out[(size_t)b * O_ + o] = e / s;
}

// ---------------------------------------------------------------------------
extern "C" void kernel_launch(void* const* d_in, const int* in_sizes, int n_in,
                              void* d_out, int out_size, void* d_ws, size_t ws_size,
                              hipStream_t stream)
{
    (void)in_sizes; (void)n_in; (void)out_size; (void)ws_size;
    const float* x     = (const float*)d_in[0];
    const float* times = (const float*)d_in[1];
    const float* W_ih  = (const float*)d_in[2];
    const float* W_hh  = (const float*)d_in[3];
    const float* b_ih  = (const float*)d_in[4];
    const float* b_hh  = (const float*)d_in[5];
    const float* w1    = (const float*)d_in[6];
    const float* b1    = (const float*)d_in[7];
    const float* w2    = (const float*)d_in[8];
    const float* b2    = (const float*)d_in[9];
    const float* w3    = (const float*)d_in[10];
    const float* b3    = (const float*)d_in[11];
    const float* Wo    = (const float*)d_in[12];
    const float* bo    = (const float*)d_in[13];

    char* base = (char*)d_ws;
    size_t off = 0;
    auto alloc = [&](size_t nbytes) -> void* {
        void* p = base + off;
        off = (off + nbytes + 255) & ~(size_t)255;   // >=32B alignment for v16h loads
        return p;
    };
    _Float16* Wih_h  = (_Float16*)alloc((size_t)3 * H_ * C_ * 2);
    _Float16* Whh_h  = (_Float16*)alloc((size_t)3 * H_ * H_ * 2);
    _Float16* w1h    = (_Float16*)alloc((size_t)HH_ * H_ * 2);
    _Float16* w2h    = (_Float16*)alloc((size_t)HH_ * HH_ * 2);
    _Float16* w3h    = (_Float16*)alloc((size_t)H_ * HH_ * 2);
    _Float16* xh     = (_Float16*)alloc((size_t)B_ * T_ * C_ * 2);
    float*    hf     = (float*)   alloc((size_t)B_ * H_ * 4);   // f32 master state
    _Float16* hzero  = (_Float16*)alloc((size_t)B_ * H_ * 2);   // all-zero h for t=0
    _Float16* hG16   = (_Float16*)alloc((size_t)B_ * H_ * 2);   // f16 h after GRU
    _Float16* hRK16  = (_Float16*)alloc((size_t)B_ * H_ * 2);   // f16 h after RK4
    _Float16* zh     = (_Float16*)alloc((size_t)B_ * H_ * 2);   // RK4 stage arg
    _Float16* a1h    = (_Float16*)alloc((size_t)B_ * HH_ * 2);
    _Float16* a2h    = (_Float16*)alloc((size_t)B_ * HH_ * 2);
    float*    k1     = (float*)   alloc((size_t)B_ * H_ * 4);
    float*    k2     = (float*)   alloc((size_t)B_ * H_ * 4);
    float*    k3     = (float*)   alloc((size_t)B_ * H_ * 4);

    auto cvt = [&](const float* s, _Float16* d, int n) {
        k_cvt_f32_f16<<<(n + 255) / 256, 256, 0, stream>>>(s, d, n);
    };
    cvt(W_ih, Wih_h, 3 * H_ * C_);
    cvt(W_hh, Whh_h, 3 * H_ * H_);
    cvt(w1, w1h, HH_ * H_);
    cvt(w2, w2h, HH_ * HH_);
    cvt(w3, w3h, H_ * HH_);
    cvt(x, xh, B_ * T_ * C_);
    k_init_h<<<(B_ * H_ + 255) / 256, 256, 0, stream>>>(hf, hzero, B_ * H_);

    auto gemmT = [&](const _Float16* A, int lda, const _Float16* W, const float* bias,
                     _Float16* Out, int N, int K) {
        const int blocks = ((B_ / 16) * (N / 64) + 7) / 8;
        gemm_tanh<<<blocks, 256, 0, stream>>>(A, lda, W, bias, Out, N, B_, N, K);
    };
    const int l3blocks = ((B_ / 16) * (H_ / 64) + 7) / 8;   // 8
    auto lin3 = [&](int mode, int t) {
        if (mode == 1)      gemm_lin3<1><<<l3blocks, 256, 0, stream>>>(a2h, w3h, b3, hf, k1, k2, k3, zh,    times, t);
        else if (mode == 2) gemm_lin3<2><<<l3blocks, 256, 0, stream>>>(a2h, w3h, b3, hf, k1, k2, k3, zh,    times, t);
        else if (mode == 3) gemm_lin3<3><<<l3blocks, 256, 0, stream>>>(a2h, w3h, b3, hf, k1, k2, k3, zh,    times, t);
        else                gemm_lin3<4><<<l3blocks, 256, 0, stream>>>(a2h, w3h, b3, hf, k1, k2, k3, hRK16, times, t);
    };
    // ODE func body (lin1 -> lin2), from a given f16 input
    auto f12 = [&](const _Float16* yin) {
        gemmT(yin, H_,  w1h, b1, a1h, HH_, H_);
        gemmT(a1h, HH_, w2h, b2, a2h, HH_, HH_);
    };
    const int grublocks = ((B_ / 16) * (H_ / 16) + 7) / 8;  // 32
    auto gru = [&](int t, const _Float16* hin) {
        gru_fused<<<grublocks, 256, 0, stream>>>(xh + (size_t)t * C_, T_ * C_,
                                                 Wih_h, Whh_h, b_ih, b_hh,
                                                 hin, hf, hG16);
    };

    // t = 0: h0 = 0
    gru(0, hzero);
    for (int t = 1; t < T_; ++t) {
        f12(hG16); lin3(1, t);     // k1, z1
        f12(zh);   lin3(2, t);     // k2, z2
        f12(zh);   lin3(3, t);     // k3, z3
        f12(zh);   lin3(4, t);     // k4, y update -> hf + hRK16
        gru(t, hRK16);             // GRU -> hf + hG16
    }
    k_head<<<B_, 32, 0, stream>>>(hf, Wo, bo, (float*)d_out);
}
